// RegionProposalNetwork_62208306315361
// MI455X (gfx1250) — matrix operand order repaired and checked by reading, not compiled, not run on hardware
//
#include <hip/hip_runtime.h>
#include <hip/hip_bf16.h>
#include <math.h>

// ---------------------------------------------------------------------------
// RPN (Faster R-CNN region proposal network) for MI455X / gfx1250.
// 3x3 conv 512->512 (94 GFLOP) + 1x1 heads as bf16 WMMA implicit GEMMs.
// Conv kernel: A tiles via CDNA5 async global->LDS copies (ASYNCcnt),
// B (weight) tiles via the Tensor Data Mover (tensor_load_to_lds, TENSORcnt)
// with hardware LDS padding matching the 80B bank-conflict-free row stride.
// Proposal stage: one 1024-thread block per image, 32768-key bitonic sort in
// 256KB LDS (CDNA5: 320KB/WGP), greedy NMS.
// ---------------------------------------------------------------------------

#define BATCH     8
#define CIN       512
#define COUT      512
#define HH        50
#define WW        50
#define HWPIX     2500                 // 50*50
#define MTOT      20000                // BATCH*HWPIX
#define NANCH     9
#define NBOX      22500                // HWPIX*NANCH
#define NSORT     32768
#define NPRE      6000
#define NPOST     300

// d_out float offsets
#define OUT_LOCS0    0                 // 22500*4
#define OUT_SCORES0  90000             // 22500*2
#define OUT_ROIS     135000            // 8*300*4
#define OUT_ROIIDX   144600            // 8*300 (int32 bits)
#define OUT_ANCHORS  147000            // 22500*4

// LDS layout for the GEMM kernels (bytes). Tile = 64 rows x 32 halfs, rows
// padded to 80B (40 halfs) so 16-row fragment reads hit distinct banks.
#define TILE_BYTES   5120              // 64*80
#define A_OFF(p)     ((p) * TILE_BYTES)
#define B_OFF(p)     (2 * TILE_BYTES + (p) * TILE_BYTES)
#define GEMM_LDS     (4 * TILE_BYTES)  // 20480 B

typedef __attribute__((ext_vector_type(16))) __bf16        v16bf;
typedef __attribute__((ext_vector_type(8)))  float         v8f;
typedef __attribute__((ext_vector_type(4)))  unsigned int  uint32x4;
typedef __attribute__((ext_vector_type(8)))  unsigned int  uint32x8;

union FragBF16 { uint4 q[2]; v16bf v; };

__device__ __forceinline__ unsigned short f2bf(float f) {
    unsigned int u = __float_as_uint(f);
    u = (u + 0x7FFFu + ((u >> 16) & 1u)) >> 16;   // round-to-nearest-even
    return (unsigned short)u;
}

// CDNA5 direct global->LDS async copy (16B per lane), tracked by ASYNCcnt.
__device__ __forceinline__ void async_g2l_b128(unsigned lds_byte, const void* gaddr) {
    asm volatile("global_load_async_to_lds_b128 %0, %1, off"
                 :: "v"(lds_byte), "v"((unsigned long long)gaddr)
                 : "memory");
}
__device__ __forceinline__ void wait_async0() {
    asm volatile("s_wait_asynccnt 0x0" ::: "memory");
}

// CDNA5 Tensor Data Mover: DMA a 2D tile (rows x 32 bf16, row stride 512
// elements) from global into LDS, hardware-padding each 64B row by 16B so the
// LDS image lands on the 80B padded layout. Wave-level op, TENSORcnt-tracked.
__device__ __forceinline__ void tdm_load_tile_b(unsigned lds_byte,
                                                const void* gaddr) {
    const unsigned long long ga = (unsigned long long)gaddr;
    uint32x4 g0;
    g0[0] = 1u;                                    // count=1 valid descriptor
    g0[1] = lds_byte;                              // lds_addr
    g0[2] = (unsigned)(ga & 0xFFFFFFFFu);          // global_addr[31:0]
    g0[3] = (unsigned)((ga >> 32) & 0x01FFFFFFu)   // global_addr[56:32]
          | (2u << 30);                            // type=2 ("image")
    uint32x8 g1;
    g1[0] = (1u << 16)                             // data_size = 2 bytes
          | (1u << 20)                             // pad_enable
          | (3u << 22)                             // pad_interval: 16 DWORDs
          | (3u << 25);                            // pad_amount:   4 DWORDs
    g1[1] = 0u;                                    // no atomic barrier
    g1[2] = 0x4000u;                               // tensor_dim0 = 1<<30 (hi)
    g1[3] = 0x4000u | (32u << 16);                 // tensor_dim1 hi, tile_dim0=32
    g1[4] = 64u;                                   // tile_dim1=64, tile_dim2=0
    g1[5] = 512u;                                  // tensor_dim0_stride = 512
    g1[6] = 0u;
    g1[7] = 0u;
    asm volatile("tensor_load_to_lds %0, %1" :: "s"(g0), "s"(g1) : "memory");
}
__device__ __forceinline__ void wait_tensor0() {
    __builtin_amdgcn_s_wait_tensorcnt((unsigned short)0);
}

// ---------------------------------------------------------------------------
// Conversions
// ---------------------------------------------------------------------------
__global__ void rpn_cvt_x(const float* __restrict__ x, unsigned short* __restrict__ xb) {
    int i = blockIdx.x * blockDim.x + threadIdx.x;
    if (i >= BATCH * CIN * HWPIX) return;
    int hw = i % HWPIX; int t = i / HWPIX;
    int c  = t % CIN;   int b = t / CIN;
    // NCHW fp32 -> NHWC bf16
    xb[((size_t)(b * HWPIX + hw)) * CIN + c] = f2bf(x[i]);
}

__global__ void rpn_cvt_w1(const float* __restrict__ w, unsigned short* __restrict__ WcT) {
    int i = blockIdx.x * blockDim.x + threadIdx.x;
    if (i >= 9 * COUT * CIN) return;
    int c = i % CIN; int t = i / CIN;
    int n = t % COUT; int r = t / COUT;
    // conv1_w (O,I,3,3) -> WcT[r][n][c]
    WcT[i] = f2bf(w[(size_t)n * CIN * 9 + (size_t)c * 9 + r]);
}

__global__ void rpn_cvt_w2(const float* __restrict__ loc_w, const float* __restrict__ score_w,
                           unsigned short* __restrict__ W2T) {
    int i = blockIdx.x * blockDim.x + threadIdx.x;
    if (i >= 64 * CIN) return;
    int c = i % CIN; int n = i / CIN;
    float v = 0.f;
    if (n < 36)      v = loc_w[(size_t)n * CIN + c];
    else if (n < 54) v = score_w[(size_t)(n - 36) * CIN + c];
    W2T[i] = f2bf(v);
}

// ---------------------------------------------------------------------------
// 3x3 conv, implicit GEMM: M=20000 pixels, N=512, K=9*512.
// 128 threads = 4 wave32 waves; 64x64 tile; double-buffered LDS; A tiles via
// per-lane async copies, B tiles via TDM; bf16 WMMA with fused bias+ReLU.
// ---------------------------------------------------------------------------
__global__ __launch_bounds__(128)
void rpn_conv3x3_wmma(const unsigned short* __restrict__ xb,
                      const unsigned short* __restrict__ WcT,
                      const float* __restrict__ bias,
                      unsigned short* __restrict__ h) {
    extern __shared__ unsigned char rpn_smem[];

    const int tid  = threadIdx.x;
    const int wv   = tid >> 5;
    const int lane = tid & 31;
    const int g    = lane >> 4;        // half-wave group (K split)
    const int lr   = lane & 15;
    const int m0   = blockIdx.x * 64;
    const int n0   = blockIdx.y * 64;

    v8f acc[4];
#pragma unroll
    for (int j = 0; j < 4; ++j)
#pragma unroll
        for (int e = 0; e < 8; ++e) acc[j][e] = 0.f;

    const int lrow = tid >> 1, lhalf = tid & 1;   // cooperative A-tile loads
    const int m = m0 + lrow;
    const bool mvalid = (m < MTOT);
    int bimg = 0, py = 0, px = 0;
    if (mvalid) { bimg = m / HWPIX; int hw = m % HWPIX; py = hw / WW; px = hw % WW; }

    // Per-lane async copies for the A (im2col) tile of K-step t into buffer p.
    auto issue_a = [&](int t, int p) {
        const int r  = t >> 4;            // 0..8   (filter tap)
        const int cs = (t & 15) << 5;     // 0..480 (input-channel chunk)
        const int dy = r / 3 - 1, dx = r % 3 - 1;
        const int ya = py + dy, xa = px + dx;
        const bool avalid = mvalid && (ya >= 0) && (ya < HH) && (xa >= 0) && (xa < WW);
        const unsigned aB = (unsigned)(A_OFF(p) + lrow * 80 + lhalf * 32);
        if (avalid) {
            const unsigned short* gA =
                xb + (((size_t)bimg * HWPIX + (size_t)ya * WW + xa) * CIN + cs + lhalf * 16);
            async_g2l_b128(aB,       gA);
            async_g2l_b128(aB + 16u, gA + 8);
        } else {
            uint4 z; z.x = z.y = z.z = z.w = 0u;
            *(uint4*)(rpn_smem + aB)      = z;
            *(uint4*)(rpn_smem + aB + 16) = z;
        }
    };
    // Wave-level TDM load for the B (weight) tile of K-step t into buffer p.
    auto issue_b = [&](int t, int p) {
        const int r  = t >> 4;
        const int cs = (t & 15) << 5;
        tdm_load_tile_b((unsigned)B_OFF(p),
                        WcT + (((size_t)r * COUT + n0) * CIN + cs));
    };

    if (wv == 0) issue_b(0, 0);
    issue_a(0, 0);
    wait_async0();
    if (wv == 0) wait_tensor0();
    __syncthreads();

    for (int t = 0; t < 144; ++t) {
        const int p = t & 1;
        if (t + 1 < 144) {
            if (wv == 0) issue_b(t + 1, p ^ 1);
            issue_a(t + 1, p ^ 1);
        }

        // A fragment 16x32: lanes 0-15 K={0..7,16..23}, lanes 16-31 +8
        const unsigned char* abase = rpn_smem + A_OFF(p) + (wv * 16 + lr) * 80;
        FragBF16 fa;
        fa.q[0] = *(const uint4*)(abase + 16 * g);
        fa.q[1] = *(const uint4*)(abase + 32 + 16 * g);
#pragma unroll
        for (int j = 0; j < 4; ++j) {
            // B fragment 32x16: lane holds column n, K = {8g..8g+7, 16+8g..16+8g+7}
            const unsigned char* bbase = rpn_smem + B_OFF(p) + (j * 16 + lr) * 80;
            FragBF16 fb;
            fb.q[0] = *(const uint4*)(bbase + 16 * g);
            fb.q[1] = *(const uint4*)(bbase + 32 + 16 * g);
            acc[j] = __builtin_amdgcn_wmma_f32_16x16x32_bf16(
                false, fa.v, false, fb.v, (short)0, acc[j], false, false);
        }
        wait_async0();
        if (wv == 0) wait_tensor0();
        __syncthreads();
    }

    // Epilogue: bias + ReLU -> bf16 h[m][n]
#pragma unroll
    for (int j = 0; j < 4; ++j) {
        const int nc = n0 + j * 16 + lr;
        const float bv = bias[nc];
#pragma unroll
        for (int v = 0; v < 8; ++v) {
            const int mm = m0 + wv * 16 + v + 8 * g;
            if (mm < MTOT) {
                float val = acc[j][v] + bv;
                val = val > 0.f ? val : 0.f;
                h[(size_t)mm * COUT + nc] = f2bf(val);
            }
        }
    }
}

// ---------------------------------------------------------------------------
// Head GEMM: M=20000, K=512, N=64 (36 loc + 18 score + pad); async-fed LDS.
// ---------------------------------------------------------------------------
__global__ __launch_bounds__(128)
void rpn_head_wmma(const unsigned short* __restrict__ h,
                   const unsigned short* __restrict__ W2T,
                   const float* __restrict__ loc_b,
                   const float* __restrict__ score_b,
                   float* __restrict__ locs,
                   float* __restrict__ scores,
                   float* __restrict__ out) {
    extern __shared__ unsigned char rpn_smem[];

    const int tid  = threadIdx.x;
    const int wv   = tid >> 5;
    const int lane = tid & 31;
    const int g    = lane >> 4;
    const int lr   = lane & 15;
    const int m0   = blockIdx.x * 64;

    v8f acc[4];
#pragma unroll
    for (int j = 0; j < 4; ++j)
#pragma unroll
        for (int e = 0; e < 8; ++e) acc[j][e] = 0.f;

    const int lrow = tid >> 1, lhalf = tid & 1;
    const int m = m0 + lrow;
    const bool mvalid = (m < MTOT);
    const unsigned short* asrc = h + ((size_t)(mvalid ? m : 0) * COUT + lhalf * 16);
    const unsigned short* bsrc = W2T + ((size_t)lrow * CIN + lhalf * 16);

    auto issue_tile = [&](int t, int p) {
        const int cs = t << 5;
        const unsigned aB = (unsigned)(A_OFF(p) + lrow * 80 + lhalf * 32);
        const unsigned bB = (unsigned)(B_OFF(p) + lrow * 80 + lhalf * 32);
        async_g2l_b128(bB,       bsrc + cs);
        async_g2l_b128(bB + 16u, bsrc + cs + 8);
        if (mvalid) {
            async_g2l_b128(aB,       asrc + cs);
            async_g2l_b128(aB + 16u, asrc + cs + 8);
        } else {
            uint4 z; z.x = z.y = z.z = z.w = 0u;
            *(uint4*)(rpn_smem + aB)      = z;
            *(uint4*)(rpn_smem + aB + 16) = z;
        }
    };

    issue_tile(0, 0);
    wait_async0();
    __syncthreads();

    for (int t = 0; t < 16; ++t) {
        const int p = t & 1;
        if (t + 1 < 16) issue_tile(t + 1, p ^ 1);

        const unsigned char* abase = rpn_smem + A_OFF(p) + (wv * 16 + lr) * 80;
        FragBF16 fa;
        fa.q[0] = *(const uint4*)(abase + 16 * g);
        fa.q[1] = *(const uint4*)(abase + 32 + 16 * g);
#pragma unroll
        for (int j = 0; j < 4; ++j) {
            const unsigned char* bbase = rpn_smem + B_OFF(p) + (j * 16 + lr) * 80;
            FragBF16 fb;
            fb.q[0] = *(const uint4*)(bbase + 16 * g);
            fb.q[1] = *(const uint4*)(bbase + 32 + 16 * g);
            acc[j] = __builtin_amdgcn_wmma_f32_16x16x32_bf16(
                false, fa.v, false, fb.v, (short)0, acc[j], false, false);
        }
        wait_async0();
        __syncthreads();
    }

#pragma unroll
    for (int j = 0; j < 4; ++j) {
        const int nc = j * 16 + lr;
#pragma unroll
        for (int v = 0; v < 8; ++v) {
            const int mm = m0 + wv * 16 + v + 8 * g;
            if (mm >= MTOT) continue;
            float val = acc[j][v];
            if (nc < 36) {
                val += loc_b[nc];
                locs[(size_t)mm * 36 + nc] = val;
                if (mm < HWPIX) out[OUT_LOCS0 + (size_t)mm * 36 + nc] = val;
            } else if (nc < 54) {
                const int q = nc - 36;
                val += score_b[q];
                scores[(size_t)mm * 18 + q] = val;
                if (mm < HWPIX) out[OUT_SCORES0 + (size_t)mm * 18 + q] = val;
            }
        }
    }
}

// ---------------------------------------------------------------------------
// Softmax foreground prob: fg = e^s1/(e^s0+e^s1) = sigmoid(s1-s0)
// ---------------------------------------------------------------------------
__global__ void rpn_fg(const float* __restrict__ scores, float* __restrict__ fg) {
    int i = blockIdx.x * blockDim.x + threadIdx.x;
    if (i >= BATCH * NBOX) return;
    float s0 = scores[(size_t)i * 2];
    float s1 = scores[(size_t)i * 2 + 1];
    fg[i] = 1.f / (1.f + __expf(s0 - s1));
}

// ---------------------------------------------------------------------------
// Anchors
// ---------------------------------------------------------------------------
__global__ void rpn_anchors(float* __restrict__ out) {
    int i = blockIdx.x * blockDim.x + threadIdx.x;
    if (i >= NBOX) return;
    int hw = i / NANCH, a = i % NANCH;
    int ri = a / 3, si = a % 3;
    const float ratios[3] = {0.5f, 1.f, 2.f};
    const float scales[3] = {8.f, 16.f, 32.f};
    float r = ratios[ri], s = scales[si];
    float ah = 16.f * s * sqrtf(r);
    float aw = 16.f * s * sqrtf(1.f / r);
    float ys = (float)((hw / WW) * 16);
    float xs = (float)((hw % WW) * 16);
    float* o = out + OUT_ANCHORS + (size_t)i * 4;
    o[0] = 8.f - 0.5f * ah + ys;
    o[1] = 8.f - 0.5f * aw + xs;
    o[2] = 8.f + 0.5f * ah + ys;
    o[3] = 8.f + 0.5f * aw + xs;
}

// ---------------------------------------------------------------------------
// Proposal: decode+clip+filter, 32768-key bitonic sort in 256KB LDS,
// greedy NMS (head-pointer scan), pack 300. One block per image.
// ---------------------------------------------------------------------------
__global__ __launch_bounds__(1024)
void rpn_proposal(const float* __restrict__ anchors_out,
                  const float* __restrict__ locs,
                  const float* __restrict__ fg,
                  float* __restrict__ roisbuf,
                  float* __restrict__ out,
                  const int* __restrict__ p_imh,
                  const int* __restrict__ p_imw,
                  const int* __restrict__ p_scale) {
    extern __shared__ unsigned char rpn_smem[];
    float* keys = (float*)rpn_smem;                       // 32768 f32
    int*   vals = (int*)(rpn_smem + (size_t)NSORT * 4);   // 32768 i32
    __shared__ int sh_cur, sh_kept, sh_head;

    const int b = blockIdx.x, tid = threadIdx.x;
    const float ih = (float)p_imh[0], iw = (float)p_imw[0];
    const float minsz = 16.f * (float)p_scale[0];
    const float NEGINF = -__builtin_inff();

    // Phase 1: decode boxes, score masking
    for (int i = tid; i < NSORT; i += 1024) {
        float s = NEGINF;
        if (i < NBOX) {
            const float* an = anchors_out + OUT_ANCHORS + (size_t)i * 4;
            float a0 = an[0], a1 = an[1], a2 = an[2], a3 = an[3];
            const float* lp = locs + (size_t)b * (NBOX * 4) + (size_t)i * 4;
            float l0 = lp[0], l1 = lp[1], l2 = lp[2], l3 = lp[3];
            float bh = a2 - a0, bw = a3 - a1;
            float cy = a0 + 0.5f * bh, cx = a1 + 0.5f * bw;
            float cty = l0 * bh + cy, ctx = l1 * bw + cx;
            float th = __expf(l2) * bh, tw = __expf(l3) * bw;
            float y1 = cty - 0.5f * th, x1 = ctx - 0.5f * tw;
            float y2 = cty + 0.5f * th, x2 = ctx + 0.5f * tw;
            y1 = fminf(fmaxf(y1, 0.f), ih); y2 = fminf(fmaxf(y2, 0.f), ih);
            x1 = fminf(fmaxf(x1, 0.f), iw); x2 = fminf(fmaxf(x2, 0.f), iw);
            float* rb = roisbuf + ((size_t)b * NBOX + i) * 4;
            rb[0] = y1; rb[1] = x1; rb[2] = y2; rb[3] = x2;
            bool valid = (y2 - y1 >= minsz) && (x2 - x1 >= minsz);
            s = valid ? fg[(size_t)b * NBOX + i] : NEGINF;
        }
        keys[i] = s;
        vals[i] = i;
    }
    __syncthreads();

    // Phase 2: bitonic sort, descending by key
    for (unsigned k = 2; k <= NSORT; k <<= 1) {
        for (unsigned j = k >> 1; j > 0; j >>= 1) {
            for (unsigned i = tid; i < NSORT; i += 1024) {
                unsigned p = i ^ j;
                if (p > i) {
                    bool desc = ((i & k) == 0);
                    float ki = keys[i], kp = keys[p];
                    bool sw = desc ? (ki < kp) : (ki > kp);
                    if (sw) {
                        keys[i] = kp; keys[p] = ki;
                        int t = vals[i]; vals[i] = vals[p]; vals[p] = t;
                    }
                }
            }
            __syncthreads();
        }
    }

    // Phase 3: gather top-6000 boxes; alias LDS (keys->boxes/areas, vals->sup/kept)
    int   myidx[6];
    float mykey[6];
    int cnt = 0;
    for (int i = tid; i < NPRE; i += 1024) { myidx[cnt] = vals[i]; mykey[cnt] = keys[i]; ++cnt; }
    __syncthreads();

    float* boxes = keys;              // [6000*4]
    float* areas = keys + 24000;      // [6000]
    int*   sup   = vals;              // [6000]
    int*   kept  = vals + 8000;       // [300]

    cnt = 0;
    for (int i = tid; i < NPRE; i += 1024) {
        int idx = myidx[cnt]; float kk = mykey[cnt]; ++cnt;
        float b0 = 0.f, b1 = 0.f, b2 = 0.f, b3 = 0.f;
        int su = 1;
        if (kk != NEGINF && idx < NBOX) {
            const float* rb = roisbuf + ((size_t)b * NBOX + idx) * 4;
            b0 = rb[0]; b1 = rb[1]; b2 = rb[2]; b3 = rb[3];
            su = 0;
        }
        boxes[i * 4 + 0] = b0; boxes[i * 4 + 1] = b1;
        boxes[i * 4 + 2] = b2; boxes[i * 4 + 3] = b3;
        areas[i] = (b2 - b0) * (b3 - b1);
        sup[i] = su;
    }
    if (tid == 0) { sh_head = 0; sh_kept = 0; }
    __syncthreads();

    // Phase 4: greedy NMS
    while (true) {
        if (tid == 0) {
            int hpos = sh_head;
            while (hpos < NPRE && sup[hpos]) hpos++;
            if (hpos >= NPRE || sh_kept >= NPOST) {
                sh_cur = -1;
            } else {
                sh_cur = hpos;
                kept[sh_kept] = hpos;
                sh_kept = sh_kept + 1;
                sup[hpos] = 1;
                hpos++;
            }
            sh_head = hpos;
        }
        __syncthreads();
        int cur = sh_cur;
        if (cur < 0) break;
        float cy1 = boxes[cur * 4 + 0], cx1 = boxes[cur * 4 + 1];
        float cy2 = boxes[cur * 4 + 2], cx2 = boxes[cur * 4 + 3];
        float ca = areas[cur];
        for (int i = cur + 1 + tid; i < NPRE; i += 1024) {
            if (!sup[i]) {
                float yy1 = fmaxf(cy1, boxes[i * 4 + 0]);
                float xx1 = fmaxf(cx1, boxes[i * 4 + 1]);
                float yy2 = fminf(cy2, boxes[i * 4 + 2]);
                float xx2 = fminf(cx2, boxes[i * 4 + 3]);
                float inter = fmaxf(yy2 - yy1, 0.f) * fmaxf(xx2 - xx1, 0.f);
                float iou = inter / (ca + areas[i] - inter);
                if (iou > 0.7f) sup[i] = 1;
            }
        }
        __syncthreads();
    }

    // Phase 5: pack (zero-padded), roi_idx
    for (int i = tid; i < NPOST * 4; i += 1024)
        out[OUT_ROIS + (size_t)b * (NPOST * 4) + i] = 0.f;
    for (int i = tid; i < NPOST; i += 1024)
        ((int*)out)[OUT_ROIIDX + (size_t)b * NPOST + i] = b;
    __syncthreads();
    const int kc = sh_kept;
    for (int k2 = tid; k2 < kc; k2 += 1024) {
        int cur = kept[k2];
        float* o = out + OUT_ROIS + (size_t)b * (NPOST * 4) + (size_t)k2 * 4;
        o[0] = boxes[cur * 4 + 0]; o[1] = boxes[cur * 4 + 1];
        o[2] = boxes[cur * 4 + 2]; o[3] = boxes[cur * 4 + 3];
    }
}

// ---------------------------------------------------------------------------
// Launch
// ---------------------------------------------------------------------------
extern "C" void kernel_launch(void* const* d_in, const int* in_sizes, int n_in,
                              void* d_out, int out_size, void* d_ws, size_t ws_size,
                              hipStream_t stream) {
    (void)in_sizes; (void)n_in; (void)out_size; (void)ws_size;
    const float* x       = (const float*)d_in[0];
    const float* conv1_w = (const float*)d_in[1];
    const float* conv1_b = (const float*)d_in[2];
    const float* score_w = (const float*)d_in[3];
    const float* score_b = (const float*)d_in[4];
    const float* loc_w   = (const float*)d_in[5];
    const float* loc_b   = (const float*)d_in[6];
    const int*   img_h   = (const int*)d_in[7];
    const int*   img_w   = (const int*)d_in[8];
    const int*   scale   = (const int*)d_in[9];
    float* out = (float*)d_out;

    unsigned char* ws = (unsigned char*)d_ws;
    size_t off = 0;
    auto take = [&](size_t bytes) -> unsigned char* {
        unsigned char* p = ws + off;
        off = (off + bytes + 255) & ~(size_t)255;
        return p;
    };
    unsigned short* xb      = (unsigned short*)take((size_t)BATCH * HWPIX * CIN * 2);   // 20.5 MB
    unsigned short* WcT     = (unsigned short*)take((size_t)9 * COUT * CIN * 2);        // 4.7 MB
    unsigned short* W2T     = (unsigned short*)take((size_t)64 * CIN * 2);              // 64 KB
    unsigned short* hbuf    = (unsigned short*)take((size_t)MTOT * COUT * 2);           // 20.5 MB
    float* locs_ws   = (float*)take((size_t)BATCH * NBOX * 4 * 4);                      // 2.9 MB
    float* scores_ws = (float*)take((size_t)BATCH * NBOX * 2 * 4);                      // 1.4 MB
    float* fg_ws     = (float*)take((size_t)BATCH * NBOX * 4);                          // 0.7 MB
    float* roisbuf   = (float*)take((size_t)BATCH * NBOX * 4 * 4);                      // 2.9 MB

    rpn_cvt_x<<<(BATCH * CIN * HWPIX + 255) / 256, 256, 0, stream>>>(x, xb);
    rpn_cvt_w1<<<(9 * COUT * CIN + 255) / 256, 256, 0, stream>>>(conv1_w, WcT);
    rpn_cvt_w2<<<(64 * CIN + 255) / 256, 256, 0, stream>>>(loc_w, score_w, W2T);

    dim3 gconv((MTOT + 63) / 64, COUT / 64);   // 313 x 8
    rpn_conv3x3_wmma<<<gconv, 128, GEMM_LDS, stream>>>(xb, WcT, conv1_b, hbuf);

    rpn_head_wmma<<<dim3((MTOT + 63) / 64, 1), 128, GEMM_LDS, stream>>>(
        hbuf, W2T, loc_b, score_b, locs_ws, scores_ws, out);

    rpn_fg<<<(BATCH * NBOX + 255) / 256, 256, 0, stream>>>(scores_ws, fg_ws);
    rpn_anchors<<<(NBOX + 255) / 256, 256, 0, stream>>>(out);

    rpn_proposal<<<BATCH, 1024, (size_t)NSORT * 8, stream>>>(
        out, locs_ws, fg_ws, roisbuf, out, img_h, img_w, scale);
}
